// GeoformerMultiHeadAttention_66125316489308
// MI455X (gfx1250) — matrix operand order repaired and checked by reading, not compiled, not run on hardware
//
#include <hip/hip_runtime.h>
#include <hip/hip_bf16.h>

// ---------------------------------------------------------------------------
// GeoformerMultiHeadAttention, MI455X (gfx1250) wave32 + WMMA bf16-split GEMMs.
//
// Heavy work: two 65536x512x512 GEMMs on edge_attr (dk path and Wea path).
// f32 GEMM emulated with split-bf16 (hi+lo) -> 3x v_wmma_f32_16x16x32_bf16 per
// K=32 chunk, f32 accumulate (~fp32 accuracy at bf16 matrix-core rate).
// Weights are pre-transposed + pre-split into bf16 once, so B fragments are
// raw 32-byte contiguous loads (2x global_load_b128). Blocks process 32 rows
// (two 16-row sub-tiles) so every B fragment feeds two accumulator chains:
// 24 WMMA per 16 B-loads per K-step. A tiles are staged in LDS as split-bf16;
// fragments are 2x ds_load_b128 + shuffle. The apn@Wdu GEMM is algebraically
// collapsed (sum over j moved inside), so apn/dk never touch HBM.
// ---------------------------------------------------------------------------

typedef __attribute__((ext_vector_type(16))) __bf16 v16bf;
typedef __attribute__((ext_vector_type(8)))  __bf16 v8bf;
typedef __attribute__((ext_vector_type(2)))  __bf16 v2bf;
typedef __attribute__((ext_vector_type(8)))  float  v8f;

#define NN 256
#define EE 512
#define HH 8
#define DD 64
#define WELEM (EE * EE)   // 262144 elements per weight matrix

__device__ __forceinline__ v8f wmma_bf16(v16bf a, v16bf b, v8f c) {
  return __builtin_amdgcn_wmma_f32_16x16x32_bf16(false, a, false, b, (short)0, c,
                                                 false, false);
}

__device__ __forceinline__ float silu_f(float x) { return x / (1.0f + __expf(-x)); }

__device__ __forceinline__ v16bf cat8(v8bf a, v8bf b) {
  return __builtin_shufflevector(a, b, 0, 1, 2, 3, 4, 5, 6, 7, 8, 9, 10, 11, 12,
                                 13, 14, 15);
}

// ---------------------------------------------------------------------------
// Prep: WT_hi[n][k], WT_lo[n][k] = split-bf16 of W[k][n]  (transpose so that a
// B-fragment lane load = 32 contiguous bytes).
// ---------------------------------------------------------------------------
__global__ __launch_bounds__(256) void prep_weight_kernel(
    const float* __restrict__ W, __bf16* __restrict__ Whi,
    __bf16* __restrict__ Wlo) {
  int idx = blockIdx.x * 256 + threadIdx.x;  // 0..262143
  int n = idx >> 9, k = idx & 511;
  float w = W[k * EE + n];
  __bf16 h = (__bf16)w;
  Whi[idx] = h;
  Wlo[idx] = (__bf16)(w - (float)h);
}

// ---------------------------------------------------------------------------
// Kernel 1: Q,K,V = x @ W{q,k,v} + b.  One 16x16 tile per wave, 4 waves/block.
// wbf holds [Wq_hi|Wq_lo|Wk_hi|Wk_lo|Wv_hi|Wv_lo], each WELEM, transposed.
// ---------------------------------------------------------------------------
__global__ __launch_bounds__(128) void qkv_kernel(
    const float* __restrict__ x, const float* __restrict__ bq,
    const float* __restrict__ bk, const float* __restrict__ bv,
    const __bf16* __restrict__ wbf, float* __restrict__ Q,
    float* __restrict__ K, float* __restrict__ V) {
  int wave = threadIdx.x >> 5;
  int lid  = threadIdx.x & 31;
  int half = lid >> 4, r16 = lid & 15;
  int tile = blockIdx.x * 4 + wave;            // 0..1535
  int which = tile / (16 * 32);
  int t2 = tile % (16 * 32);
  int m0 = (t2 / 32) * 16, n0 = (t2 % 32) * 16;
  const float* bias = (which == 0) ? bq : (which == 1) ? bk : bv;
  float* Out        = (which == 0) ? Q  : (which == 1) ? K  : V;
  const __bf16* Whi = wbf + (size_t)which * (2 * WELEM);
  const __bf16* Wlo = Whi + WELEM;

  v8f acc = {};
  for (int k0 = 0; k0 < EE; k0 += 32) {
    v16bf ah, al;
    const float* ap0 = &x[(m0 + r16) * EE + k0 + half * 16];
#pragma unroll
    for (int e = 0; e < 8; e++) {
      float a = ap0[e];
      __bf16 h = (__bf16)a;
      ah[e] = h;
      al[e] = (__bf16)(a - (float)h);
    }
#pragma unroll
    for (int e = 0; e < 8; e++) {
      float a = ap0[8 + e];
      __bf16 h = (__bf16)a;
      ah[8 + e] = h;
      al[8 + e] = (__bf16)(a - (float)h);
    }
    size_t boff = (size_t)(n0 + r16) * EE + k0 + half * 16;
    v16bf bh = *(const v16bf*)&Whi[boff];
    v16bf bl = *(const v16bf*)&Wlo[boff];
    acc = wmma_bf16(ah, bh, acc);
    acc = wmma_bf16(ah, bl, acc);
    acc = wmma_bf16(al, bh, acc);
  }
  float bn = bias[n0 + r16];
#pragma unroll
  for (int r = 0; r < 8; r++) {
    int m = m0 + r + 8 * half;                 // C/D: row = r + 8*half, col = r16
    Out[m * EE + n0 + r16] = acc[r] + bn;
  }
}

// ---------------------------------------------------------------------------
// Kernel 1b: ksum[h][n] = sum_d K[n][h*64+d]  (einsum's k-collapse)
// ---------------------------------------------------------------------------
__global__ void ksum_kernel(const float* __restrict__ K, float* __restrict__ ksum) {
  int idx = blockIdx.x * blockDim.x + threadIdx.x;
  if (idx >= HH * NN) return;
  int h = idx >> 8, n = idx & 255;
  float s = 0.0f;
  for (int d = 0; d < DD; d++) s += K[n * EE + h * DD + d];
  ksum[h * NN + n] = s;
}

// Stage a rows x 512 f32 tile into LDS as split-bf16 (pairs -> 4-byte stores).
template <int ROWS>
__device__ __forceinline__ void stage_tile(const float* __restrict__ src,
                                           __bf16* Ahi, __bf16* Alo, int tid) {
  for (int base = tid * 2; base < ROWS * EE; base += 512) {
    float2 a = *(const float2*)&src[base];
    __bf16 h0 = (__bf16)a.x, h1 = (__bf16)a.y;
    v2bf hh = {h0, h1};
    v2bf ll = {(__bf16)(a.x - (float)h0), (__bf16)(a.y - (float)h1)};
    *(v2bf*)&Ahi[base] = hh;
    *(v2bf*)&Alo[base] = ll;
  }
}

// Lane's A-fragment: two contiguous 16B runs -> 2x ds_load_b128 + shuffle.
__device__ __forceinline__ v16bf a_frag(const __bf16* A, int r16, int half,
                                        int k0) {
  const v8bf a0 = *(const v8bf*)&A[r16 * EE + k0 + half * 16];
  const v8bf a1 = *(const v8bf*)&A[r16 * EE + k0 + half * 16 + 8];
  return cat8(a0, a1);
}

// ---------------------------------------------------------------------------
// Kernel 2: per-row-i fused attention. 256 blocks (one per i), 256 threads.
// 32 j-rows per iteration (two 16-row WMMA sub-tiles share each B fragment).
//   dkrow = silu(edge_attr[i, j0:j0+32, :] @ Wdk + bdk)  (WMMA)
//   wave w owns head w's 64 columns -> q.dk dot reduced in-wave
//   probs = silu(ksum * dot) * cosine_cutoff(dist)
//   attn[i,:] += probs * V[j,:];  P[i,c,:] += vec[i,j,c] * probs * V[j,:]
// ---------------------------------------------------------------------------
__global__ __launch_bounds__(256) void attn_kernel(
    const float* __restrict__ edge_attr, const __bf16* __restrict__ WdkT_hi,
    const __bf16* __restrict__ WdkT_lo, const float* __restrict__ bdk,
    const float* __restrict__ Q, const float* __restrict__ V,
    const float* __restrict__ ksum, const float* __restrict__ dist,
    const float* __restrict__ vec, float* __restrict__ out_attn,
    float* __restrict__ P) {
  __shared__ __bf16 Ahi[32 * EE];
  __shared__ __bf16 Alo[32 * EE];
  __shared__ float  probs[32][HH];
  __shared__ float  qi_s[EE];
  __shared__ float  vtile[32][3];

  int i = blockIdx.x;
  int tid = threadIdx.x;
  int wave = tid >> 5, lid = tid & 31, half = lid >> 4, r16 = lid & 15;
  int e0 = tid * 2;

  qi_s[tid]       = Q[i * EE + tid];
  qi_s[tid + 256] = Q[i * EE + 256 + tid];
  float ks = ksum[wave * NN + i];

  float attn0 = 0.0f, attn1 = 0.0f;
  float Pr[3][2] = {{0, 0}, {0, 0}, {0, 0}};

  for (int j0 = 0; j0 < NN; j0 += 32) {
    __syncthreads();  // covers qi_s on first iter; LDS reuse on later iters
    stage_tile<32>(&edge_attr[((size_t)(i * NN + j0)) * EE], Ahi, Alo, tid);
    if (tid < 96) {
      int jr = tid / 3, c = tid % 3;
      vtile[jr][c] = vec[((size_t)(i * NN + j0 + jr)) * 3 + c];
    }
    // prefetch next 32 rows of edge_attr (two 128B lines per thread)
    if (j0 + 32 < NN) {
      const float* nx =
          &edge_attr[((size_t)(i * NN + j0 + 32 + (tid >> 4))) * EE +
                     (tid & 15) * 32];
      __builtin_prefetch(nx, 0, 3);
      __builtin_prefetch(nx + 16 * EE, 0, 3);
    }
    __syncthreads();

    // GEMM: 32 rows x 64 cols per wave (4 N-tiles x 2 M-sub-tiles), K=512
    v8f acc[4][2];
#pragma unroll
    for (int t = 0; t < 4; t++) { acc[t][0] = {}; acc[t][1] = {}; }
    for (int k0 = 0; k0 < EE; k0 += 32) {
      v16bf ah0 = a_frag(Ahi, r16, half, k0);
      v16bf al0 = a_frag(Alo, r16, half, k0);
      v16bf ah1 = a_frag(Ahi + 16 * EE, r16, half, k0);
      v16bf al1 = a_frag(Alo + 16 * EE, r16, half, k0);
#pragma unroll
      for (int t = 0; t < 4; t++) {
        int n0 = wave * DD + t * 16;
        size_t boff = (size_t)(n0 + r16) * EE + k0 + half * 16;
        v16bf bh = *(const v16bf*)&WdkT_hi[boff];
        v16bf bl = *(const v16bf*)&WdkT_lo[boff];
        acc[t][0] = wmma_bf16(ah0, bh, acc[t][0]);
        acc[t][1] = wmma_bf16(ah1, bh, acc[t][1]);
        acc[t][0] = wmma_bf16(ah0, bl, acc[t][0]);
        acc[t][1] = wmma_bf16(ah1, bl, acc[t][1]);
        acc[t][0] = wmma_bf16(al0, bh, acc[t][0]);
        acc[t][1] = wmma_bf16(al1, bh, acc[t][1]);
      }
    }

    // epilogue: silu(dk+b) dotted with q over this wave's 64 d-columns
    float part[2][8];
#pragma unroll
    for (int s = 0; s < 2; s++)
#pragma unroll
      for (int r = 0; r < 8; r++) part[s][r] = 0.0f;
#pragma unroll
    for (int t = 0; t < 4; t++) {
      int n0 = wave * DD + t * 16;
      float bb = bdk[n0 + r16];
      float qv = qi_s[n0 + r16];
#pragma unroll
      for (int s = 0; s < 2; s++)
#pragma unroll
        for (int r = 0; r < 8; r++)
          part[s][r] += silu_f(acc[t][s][r] + bb) * qv;
    }
#pragma unroll
    for (int m = 1; m < 16; m <<= 1) {
#pragma unroll
      for (int s = 0; s < 2; s++)
#pragma unroll
        for (int r = 0; r < 8; r++) part[s][r] += __shfl_xor(part[s][r], m, 32);
    }
    if (r16 == 0) {
#pragma unroll
      for (int s = 0; s < 2; s++)
#pragma unroll
        for (int r = 0; r < 8; r++) {
          int jl = s * 16 + r + 8 * half;
          float aw = ks * part[s][r];
          float pr = silu_f(aw);
          float d = dist[i * NN + j0 + jl];
          float cut = (d < 5.0f)
                          ? 0.5f * (__cosf(d * 0.6283185307179586f) + 1.0f)
                          : 0.0f;
          probs[jl][wave] = pr * cut;
        }
    }
    __syncthreads();

    // apn accumulation: thread owns e0, e0+1 (same head since e0 is even)
    int hsel = e0 >> 6;
    for (int j = 0; j < 32; j++) {
      const float* vrow = &V[(size_t)(j0 + j) * EE];
      float p = probs[j][hsel];
      float a0 = p * vrow[e0];
      float a1 = p * vrow[e0 + 1];
      attn0 += a0;
      attn1 += a1;
      float c0 = vtile[j][0], c1 = vtile[j][1], c2 = vtile[j][2];
      Pr[0][0] += c0 * a0; Pr[0][1] += c0 * a1;
      Pr[1][0] += c1 * a0; Pr[1][1] += c1 * a1;
      Pr[2][0] += c2 * a0; Pr[2][1] += c2 * a1;
    }
  }

  out_attn[i * EE + e0]     = attn0;
  out_attn[i * EE + e0 + 1] = attn1;
#pragma unroll
  for (int c = 0; c < 3; c++) {
    P[(size_t)i * 1536 + c * EE + e0]     = Pr[c][0];
    P[(size_t)i * 1536 + c * EE + e0 + 1] = Pr[c][1];
  }
}

// ---------------------------------------------------------------------------
// Kernel 3: du = P @ Wdu + vecsum*bdu (3x512x512 per i, VALU), vec-layernorm
// (norm over c, max/min over e), then wswt = du @ Wdih -> ws | wt.
// ---------------------------------------------------------------------------
__global__ __launch_bounds__(256) void du_kernel(
    const float* __restrict__ P, const float* __restrict__ vec,
    const float* __restrict__ Wdu, const float* __restrict__ bdu,
    const float* __restrict__ Wdih, float* __restrict__ wsS,
    float* __restrict__ wtS) {
  __shared__ float Ps[3][EE];
  __shared__ float du[3][EE];
  __shared__ float nrm[EE];
  __shared__ float vsum[3];
  __shared__ float red[24];
  __shared__ float redmx[8], redmn[8];

  int i = blockIdx.x, tid = threadIdx.x;

  for (int idx = tid; idx < 3 * EE; idx += 256)
    Ps[idx / EE][idx % EE] = P[(size_t)i * 1536 + idx];

  // vecsum[c] = sum_j vec[i,j,c]
  const float* vr = &vec[(size_t)i * (NN * 3) + tid * 3];
  float v0 = vr[0], v1 = vr[1], v2 = vr[2];
#pragma unroll
  for (int m = 1; m < 32; m <<= 1) {
    v0 += __shfl_xor(v0, m, 32);
    v1 += __shfl_xor(v1, m, 32);
    v2 += __shfl_xor(v2, m, 32);
  }
  if ((tid & 31) == 0) {
    int w = tid >> 5;
    red[w] = v0; red[8 + w] = v1; red[16 + w] = v2;
  }
  __syncthreads();
  if (tid < 3) {
    float s = 0.0f;
    for (int w = 0; w < 8; w++) s += red[tid * 8 + w];
    vsum[tid] = s;
  }
  __syncthreads();

  // du and clamped norms
  for (int eo = 0; eo < 2; eo++) {
    int e = tid + eo * 256;
    float be = bdu[e];
    float s0 = vsum[0] * be, s1 = vsum[1] * be, s2 = vsum[2] * be;
    for (int k = 0; k < EE; k++) {
      float w = Wdu[k * EE + e];
      s0 += Ps[0][k] * w;
      s1 += Ps[1][k] * w;
      s2 += Ps[2][k] * w;
    }
    du[0][e] = s0; du[1][e] = s1; du[2][e] = s2;
    nrm[e] = fmaxf(sqrtf(s0 * s0 + s1 * s1 + s2 * s2), 1e-12f);
  }
  __syncthreads();

  // max/min over the 512 clamped norms
  float mx = fmaxf(nrm[tid], nrm[tid + 256]);
  float mn = fminf(nrm[tid], nrm[tid + 256]);
#pragma unroll
  for (int m = 1; m < 32; m <<= 1) {
    mx = fmaxf(mx, __shfl_xor(mx, m, 32));
    mn = fminf(mn, __shfl_xor(mn, m, 32));
  }
  if ((tid & 31) == 0) { redmx[tid >> 5] = mx; redmn[tid >> 5] = mn; }
  __syncthreads();
  if (tid == 0) {
    float M = redmx[0], Nv = redmn[0];
    for (int w = 1; w < 8; w++) { M = fmaxf(M, redmx[w]); Nv = fminf(Nv, redmn[w]); }
    redmx[0] = M; redmn[0] = Nv;
  }
  __syncthreads();
  float sMax = redmx[0], sMin = redmn[0];
  float delta = sMax - sMin;
  if (delta == 0.0f) delta = 1.0f;

  for (int eo = 0; eo < 2; eo++) {
    int e = tid + eo * 256;
    float nd = nrm[e];
    float scale = fmaxf((nd - sMin) / delta, 0.0f) / nd;  // relu(dmm)*du/nd
    du[0][e] *= scale; du[1][e] *= scale; du[2][e] *= scale;
  }
  __syncthreads();

  // wswt = du @ Wdih  ([3,512] @ [512,1024])
  for (int oo = 0; oo < 4; oo++) {
    int o = tid + oo * 256;
    float s0 = 0.0f, s1 = 0.0f, s2 = 0.0f;
    for (int k = 0; k < EE; k++) {
      float w = Wdih[k * (2 * EE) + o];
      s0 += du[0][k] * w;
      s1 += du[1][k] * w;
      s2 += du[2][k] * w;
    }
    if (o < EE) {
      wsS[(size_t)i * 1536 + 0 * EE + o] = s0;
      wsS[(size_t)i * 1536 + 1 * EE + o] = s1;
      wsS[(size_t)i * 1536 + 2 * EE + o] = s2;
    } else {
      int oe = o - EE;
      wtS[(size_t)i * 1536 + 0 * EE + oe] = s0;
      wtS[(size_t)i * 1536 + 1 * EE + oe] = s1;
      wtS[(size_t)i * 1536 + 2 * EE + oe] = s2;
    }
  }
}

// ---------------------------------------------------------------------------
// Kernel 4: ipe[i,j,e] = silu(edge_attr@Wea + bea) * sum_c ws[i,c,e]*wt[j,c,e]
// Blocks: 256 i * 8 j-tiles of 32 rows.
// ---------------------------------------------------------------------------
__global__ __launch_bounds__(256) void ipe_kernel(
    const float* __restrict__ edge_attr, const __bf16* __restrict__ WeaT_hi,
    const __bf16* __restrict__ WeaT_lo, const float* __restrict__ bea,
    const float* __restrict__ wsS, const float* __restrict__ wtS,
    float* __restrict__ out_ipe) {
  __shared__ __bf16 Ahi[32 * EE];
  __shared__ __bf16 Alo[32 * EE];

  int i  = blockIdx.x >> 3;
  int j0 = (blockIdx.x & 7) * 32;
  int tid = threadIdx.x;
  int wave = tid >> 5, lid = tid & 31, half = lid >> 4, r16 = lid & 15;

  stage_tile<32>(&edge_attr[((size_t)(i * NN + j0)) * EE], Ahi, Alo, tid);
  __syncthreads();

  v8f acc[4][2];
#pragma unroll
  for (int t = 0; t < 4; t++) { acc[t][0] = {}; acc[t][1] = {}; }
  for (int k0 = 0; k0 < EE; k0 += 32) {
    v16bf ah0 = a_frag(Ahi, r16, half, k0);
    v16bf al0 = a_frag(Alo, r16, half, k0);
    v16bf ah1 = a_frag(Ahi + 16 * EE, r16, half, k0);
    v16bf al1 = a_frag(Alo + 16 * EE, r16, half, k0);
#pragma unroll
    for (int t = 0; t < 4; t++) {
      int n0 = wave * DD + t * 16;
      size_t boff = (size_t)(n0 + r16) * EE + k0 + half * 16;
      v16bf bh = *(const v16bf*)&WeaT_hi[boff];
      v16bf bl = *(const v16bf*)&WeaT_lo[boff];
      acc[t][0] = wmma_bf16(ah0, bh, acc[t][0]);
      acc[t][1] = wmma_bf16(ah1, bh, acc[t][1]);
      acc[t][0] = wmma_bf16(ah0, bl, acc[t][0]);
      acc[t][1] = wmma_bf16(ah1, bl, acc[t][1]);
      acc[t][0] = wmma_bf16(al0, bh, acc[t][0]);
      acc[t][1] = wmma_bf16(al1, bh, acc[t][1]);
    }
  }

#pragma unroll
  for (int t = 0; t < 4; t++) {
    int e = wave * DD + t * 16 + r16;
    float bb = bea[e];
    float w0 = wsS[(size_t)i * 1536 + 0 * EE + e];
    float w1 = wsS[(size_t)i * 1536 + 1 * EE + e];
    float w2 = wsS[(size_t)i * 1536 + 2 * EE + e];
#pragma unroll
    for (int s = 0; s < 2; s++)
#pragma unroll
      for (int r = 0; r < 8; r++) {
        int j = j0 + s * 16 + r + 8 * half;
        float g = silu_f(acc[t][s][r] + bb);
        float ib = w0 * wtS[(size_t)j * 1536 + 0 * EE + e] +
                   w1 * wtS[(size_t)j * 1536 + 1 * EE + e] +
                   w2 * wtS[(size_t)j * 1536 + 2 * EE + e];
        out_ipe[((size_t)(i * NN + j)) * EE + e] = g * ib;
      }
  }
}

// ---------------------------------------------------------------------------
// Launch
// ---------------------------------------------------------------------------
extern "C" void kernel_launch(void* const* d_in, const int* in_sizes, int n_in,
                              void* d_out, int out_size, void* d_ws,
                              size_t ws_size, hipStream_t stream) {
  const float* x         = (const float*)d_in[0];
  const float* vec       = (const float*)d_in[1];
  const float* dist      = (const float*)d_in[2];
  const float* edge_attr = (const float*)d_in[3];
  const float* Wq  = (const float*)d_in[4];
  const float* bq  = (const float*)d_in[5];
  const float* Wk  = (const float*)d_in[6];
  const float* bk  = (const float*)d_in[7];
  const float* Wv  = (const float*)d_in[8];
  const float* bv  = (const float*)d_in[9];
  const float* Wdk = (const float*)d_in[10];
  const float* bdk = (const float*)d_in[11];
  const float* Wdu = (const float*)d_in[12];
  const float* bdu = (const float*)d_in[13];
  const float* Wdih= (const float*)d_in[14];
  const float* Wea = (const float*)d_in[15];
  const float* bea = (const float*)d_in[16];

  float* ws = (float*)d_ws;
  float* Q    = ws;                 // 131072
  float* K    = ws + 131072;        // 131072
  float* V    = ws + 262144;        // 131072
  float* ksum = ws + 393216;        // 2048
  float* P    = ws + 395264;        // 393216
  float* wsS  = ws + 788480;        // 393216
  float* wtS  = ws + 1181696;       // 393216  -> float end: 1574912

  // split-bf16 transposed weights: [Wq|Wk|Wv|Wdk|Wea], each hi+lo (2*WELEM)
  __bf16* wbf = (__bf16*)(ws + 1574912);        // 10*WELEM bf16 = 5 MB
  __bf16* WqkvT   = wbf;                        // 3 * 2*WELEM
  __bf16* WdkT_hi = wbf + (size_t)3 * 2 * WELEM;
  __bf16* WdkT_lo = WdkT_hi + WELEM;
  __bf16* WeaT_hi = wbf + (size_t)4 * 2 * WELEM;
  __bf16* WeaT_lo = WeaT_hi + WELEM;

  float* out_attn = (float*)d_out;            // [256,512]
  float* out_ipe  = (float*)d_out + 131072;   // [256,256,512]

  prep_weight_kernel<<<1024, 256, 0, stream>>>(Wq,  WqkvT,               WqkvT + WELEM);
  prep_weight_kernel<<<1024, 256, 0, stream>>>(Wk,  WqkvT + 2 * WELEM,   WqkvT + 3 * WELEM);
  prep_weight_kernel<<<1024, 256, 0, stream>>>(Wv,  WqkvT + 4 * WELEM,   WqkvT + 5 * WELEM);
  prep_weight_kernel<<<1024, 256, 0, stream>>>(Wdk, WdkT_hi, WdkT_lo);
  prep_weight_kernel<<<1024, 256, 0, stream>>>(Wea, WeaT_hi, WeaT_lo);

  qkv_kernel<<<384, 128, 0, stream>>>(x, bq, bk, bv, WqkvT, Q, K, V);
  ksum_kernel<<<8, 256, 0, stream>>>(K, ksum);
  attn_kernel<<<256, 256, 0, stream>>>(edge_attr, WdkT_hi, WdkT_lo, bdk, Q, V,
                                       ksum, dist, vec, out_attn, P);
  du_kernel<<<256, 256, 0, stream>>>(P, vec, Wdu, bdu, Wdih, wsS, wtS);
  ipe_kernel<<<2048, 256, 0, stream>>>(edge_attr, WeaT_hi, WeaT_lo, bea, wsS,
                                       wtS, out_ipe);
}
